// RippleAttentionBlock_2259152798577
// MI455X (gfx1250) — matrix operand order repaired and checked by reading, not compiled, not run on hardware
//
#include <hip/hip_runtime.h>
#include <hip/hip_bf16.h>
#include <math.h>

// ---------------- problem constants ----------------
#define B_   2
#define S_   4096
#define C_   1024
#define H_   8
#define D_   128
#define M_   (B_*S_)      // 8192 rows
#define N3_  (3*C_)       // 3072
#define KC_  7
#define NOFF 14           // 1 + floor(log2(4096))+1 offsets

typedef __bf16 bf16_t;
typedef __attribute__((ext_vector_type(8)))  bf16_t bf16x8;
typedef __attribute__((ext_vector_type(16))) bf16_t bf16x16;
typedef __attribute__((ext_vector_type(8)))  float  v8f;
typedef int v4i_t __attribute__((vector_size(16)));   // matches builtin param

// gfx1250 async global->LDS staging (ASYNCcnt path). Probe-confirmed present;
// parameters are AS1/AS3 pointers to 16-byte int vectors (from the compiler
// diagnostic), plus imm offset and imm cpol.
#if __has_builtin(__builtin_amdgcn_global_load_async_to_lds_b128) && \
    __has_builtin(__builtin_amdgcn_s_wait_asynccnt)
#define USE_ASYNC_LDS 1
#define GL_ASYNC_B128(gp, lp)                                        \
  __builtin_amdgcn_global_load_async_to_lds_b128(                    \
      (__attribute__((address_space(1))) v4i_t*)(gp),                \
      (__attribute__((address_space(3))) v4i_t*)(lp), 0, 0)
#else
#define USE_ASYNC_LDS 0
#endif

// ---------------- converts / transposes ----------------
__global__ __launch_bounds__(256) void k_f32_to_bf16(const float* __restrict__ in,
                                                     bf16_t* __restrict__ out, int n) {
  int i = blockIdx.x * 256 + threadIdx.x;
  int stride = gridDim.x * 256;
  for (; i < n; i += stride) out[i] = (bf16_t)in[i];
}

// W: K x N row-major f32  ->  Wt: N x K row-major bf16 (i.e. transposed)
__global__ __launch_bounds__(256) void k_transpose_bf16(const float* __restrict__ W,
                                                        bf16_t* __restrict__ Wt,
                                                        int K, int N) {
  int i = blockIdx.x * 256 + threadIdx.x;
  int total = K * N;
  int stride = gridDim.x * 256;
  for (; i < total; i += stride) {
    int n = i / K, k = i - n * K;
    Wt[i] = (bf16_t)W[(size_t)k * N + n];
  }
}

// ---------------- WMMA bf16 GEMM: C = A * Bt^T (+ Res) ----------------
// A  : M x K bf16 row-major
// Bt : N x K bf16 row-major (B transposed), so B[k][n] = Bt[n][k]
// Res: optional M x N f32 residual,  Cout: M x N f32
#define TM 128
#define TN 128
#define TK 32
#define LDST 40   // padded LDS row stride in bf16 elems (80B, 16B multiple)

__global__ __launch_bounds__(256) void k_gemm_bf16_wmma(
    const bf16_t* __restrict__ A, const bf16_t* __restrict__ Bt,
    const float* __restrict__ Res, float* __restrict__ Cout,
    int M, int N, int K)
{
  __shared__ bf16_t Al[2][TM * LDST];
  __shared__ bf16_t Bl[2][TN * LDST];

  const int tid  = threadIdx.x;
  const int lane = tid & 31;
  const int wave = tid >> 5;
  const int bm = blockIdx.x * TM;
  const int bn = blockIdx.y * TN;
  const int wm = (wave & 3) * 32;   // wave's M offset inside tile (4 strips)
  const int wn = (wave >> 2) * 64;  // wave's N offset inside tile (2 strips)

  v8f acc[2][4];
#pragma unroll
  for (int mi = 0; mi < 2; ++mi)
#pragma unroll
    for (int ni = 0; ni < 4; ++ni)
      acc[mi][ni] = (v8f){0.f,0.f,0.f,0.f,0.f,0.f,0.f,0.f};

  // CDNA5 16-bit WMMA fragment addressing (wave32):
  // A 16x32: lane holds row (lane&15); K chunks [kb..kb+7] and [kb+16..kb+23], kb=(lane>>4)*8
  // B 32x16: lane holds col (lane&15); K chunks [kb..kb+7] and [kb+16..kb+23], kb=(lane>>4)*8
  const int frow = lane & 15;
  const int fkbA = (lane >> 4) * 8;
  const int fkbB = (lane >> 4) * 8;

  auto stage = [&](int buf, int k0) {
#pragma unroll
    for (int it = 0; it < 2; ++it) {
      int slot = tid + it * 256;      // 512 slots = 128 rows x 4 chunks of 8 bf16
      int r  = slot >> 2;
      int cc = (slot & 3) * 8;
      const bf16_t* ga = A  + (size_t)(bm + r) * K + k0 + cc;
      const bf16_t* gb = Bt + (size_t)(bn + r) * K + k0 + cc;
#if USE_ASYNC_LDS
      GL_ASYNC_B128(ga, &Al[buf][r * LDST + cc]);
      GL_ASYNC_B128(gb, &Bl[buf][r * LDST + cc]);
#else
      *(bf16x8*)&Al[buf][r * LDST + cc] = *(const bf16x8*)ga;
      *(bf16x8*)&Bl[buf][r * LDST + cc] = *(const bf16x8*)gb;
      if (k0 + TK < K) {                       // hint next K tile into cache
        __builtin_prefetch(ga + TK, 0, 1);     // -> global_prefetch_b8
        __builtin_prefetch(gb + TK, 0, 1);
      }
#endif
    }
  };

  stage(0, 0);
#if USE_ASYNC_LDS
  __builtin_amdgcn_s_wait_asynccnt(0);
#endif
  __syncthreads();

  const int nk = K / TK;
#pragma unroll 1
  for (int kt = 0; kt < nk; ++kt) {
    const int buf = kt & 1;
    if (kt + 1 < nk) stage(buf ^ 1, (kt + 1) * TK);

    // gather all fragments first, then burst the 8 WMMAs back-to-back
    bf16x16 af[2], bfv[4];
#pragma unroll
    for (int mi = 0; mi < 2; ++mi) {
      const bf16_t* pa = &Al[buf][(wm + mi * 16 + frow) * LDST + fkbA];
      bf16x8 lo = *(const bf16x8*)pa;
      bf16x8 hi = *(const bf16x8*)(pa + 16);
#pragma unroll
      for (int t = 0; t < 8; ++t) { af[mi][t] = lo[t]; af[mi][t + 8] = hi[t]; }
    }
#pragma unroll
    for (int ni = 0; ni < 4; ++ni) {
      const bf16_t* pb = &Bl[buf][(wn + ni * 16 + frow) * LDST + fkbB];
      bf16x8 lo = *(const bf16x8*)pb;
      bf16x8 hi = *(const bf16x8*)(pb + 16);
#pragma unroll
      for (int t = 0; t < 8; ++t) { bfv[ni][t] = lo[t]; bfv[ni][t + 8] = hi[t]; }
    }

#pragma unroll
    for (int ni = 0; ni < 4; ++ni)
#pragma unroll
      for (int mi = 0; mi < 2; ++mi)
        acc[mi][ni] = __builtin_amdgcn_wmma_f32_16x16x32_bf16(
            /*neg_a=*/false, af[mi], /*neg_b=*/false, bfv[ni],
            /*c_mod=*/(short)0, acc[mi][ni], /*reuse_a=*/false, /*reuse_b=*/false);

#if USE_ASYNC_LDS
    __builtin_amdgcn_s_wait_asynccnt(0);
#endif
    __syncthreads();
  }

  // C/D layout: VGPR i, lanes 0-15 -> M = i, lanes 16-31 -> M = 8+i; N = lane&15
  const bool hasRes = (Res != nullptr);
#pragma unroll
  for (int mi = 0; mi < 2; ++mi) {
#pragma unroll
    for (int ni = 0; ni < 4; ++ni) {
      int rbase = bm + wm + mi * 16 + (lane >> 4) * 8;
      int col   = bn + wn + ni * 16 + (lane & 15);
#pragma unroll
      for (int i = 0; i < 8; ++i) {
        size_t idx = (size_t)(rbase + i) * N + col;
        float v = acc[mi][ni][i];
        if (hasRes) v += Res[idx];
        Cout[idx] = v;
      }
    }
  }
}

// ---------------- kv / fksum reduction (linear attention state) ----------------
// kv[b,h,d,e] = sum_s fk[b,h,s,d] * v[b,h,s,e];  fksum[b,h,d] = sum_s fk
#define KV_TILE 64
#define KV_NCH  8
#define KV_SCH  (S_ / KV_NCH)   // 512 positions per block

__global__ __launch_bounds__(256) void k_kvstats(const float* __restrict__ qkv,
                                                 float* __restrict__ kvbuf,
                                                 float* __restrict__ fksum)
{
  __shared__ float fkl[KV_TILE][D_];   // 32KB
  __shared__ float vl [KV_TILE][D_];   // 32KB
  const int tid = threadIdx.x;
  const int bh = blockIdx.x;           // 0..15
  const int b = bh >> 3, h = bh & 7;
  const int chunk = blockIdx.y;
  const int d  = tid >> 1;             // 0..127
  const int eh = (tid & 1) * 64;       // which half of e

  float acc[64];
#pragma unroll
  for (int j = 0; j < 64; ++j) acc[j] = 0.f;
  float fs = 0.f;

  for (int t = 0; t < KV_SCH / KV_TILE; ++t) {
    const int s0 = chunk * KV_SCH + t * KV_TILE;
#pragma unroll 1
    for (int it = 0; it < (KV_TILE * D_ / 256); ++it) {   // 32 iters
      int slot = tid + it * 256;
      int r  = slot >> 7;       // 0..63
      int cc = slot & 127;
      size_t base = (size_t)(b * S_ + s0 + r) * N3_;
      float kvv = qkv[base + C_ + h * D_ + cc];
      fkl[r][cc] = kvv > 0.f ? kvv + 1.f : __expf(kvv);   // elu(k)+1
      vl [r][cc] = qkv[base + 2 * C_ + h * D_ + cc];
    }
    __syncthreads();
#pragma unroll 1
    for (int ss = 0; ss < KV_TILE; ++ss) {
      float f = fkl[ss][d];
      fs += f;
      const float* vv = &vl[ss][eh];
#pragma unroll
      for (int j = 0; j < 64; ++j) acc[j] += f * vv[j];
    }
    __syncthreads();
  }

  float* kvp = kvbuf + ((size_t)bh * D_ + d) * D_ + eh;
  for (int j = 0; j < 64; ++j) atomicAdd(&kvp[j], acc[j]);
  if (eh == 0) atomicAdd(&fksum[(size_t)bh * D_ + d], fs);
}

// ---------------- fused ripple + conv + linear attention ----------------
// one block per (b,s); 256 threads, 4 channels each; writes bf16 h.
__global__ __launch_bounds__(256) void k_fuse(const float* __restrict__ qkv,
                                              const float* __restrict__ kvbuf,
                                              const float* __restrict__ fksum,
                                              const float* __restrict__ conv_w,
                                              bf16_t* __restrict__ hb)
{
  __shared__ float fq[C_];
  __shared__ float sc [H_][NOFF];
  __shared__ float wgt[H_][NOFF];
  __shared__ float zz [H_];

  const int tid = threadIdx.x;
  const int b = blockIdx.x / S_;
  const int s = blockIdx.x - b * S_;
  const float* qrow = qkv + (size_t)(b * S_ + s) * N3_;

  {   // fq = elu(q)+1 into LDS
    int c0 = tid * 4;
#pragma unroll
    for (int j = 0; j < 4; ++j) {
      float q = qrow[c0 + j];
      fq[c0 + j] = q > 0.f ? q + 1.f : __expf(q);
    }
  }
  __syncthreads();

  if (tid < H_ * NOFF) {   // ripple scores: q . k[s-off] / sqrt(D)
    int h = tid / NOFF, kk = tid - h * NOFF;
    int off = (kk == 0) ? 0 : (1 << (kk - 1));
    int pos = s - off;
    float scv = -1e30f;
    if (pos >= 0) {
      const float* kr = qkv + (size_t)(b * S_ + pos) * N3_ + C_ + h * D_;
      const float* qh = qrow + h * D_;
      float a = 0.f;
#pragma unroll 4
      for (int dd = 0; dd < D_; ++dd) a += qh[dd] * kr[dd];
      scv = a * 0.088388347648318447f;   // 1/sqrt(128)
    }
    sc[h][kk] = scv;
  }
  __syncthreads();

  if (tid < H_) {          // per-head softmax over 14 + z = fq . fksum
    int h = tid;
    float m = -1e30f;
    for (int k = 0; k < NOFF; ++k) m = fmaxf(m, sc[h][k]);
    float ssum = 0.f;
    for (int k = 0; k < NOFF; ++k) {
      float e = (sc[h][k] <= -1e29f) ? 0.f : __expf(sc[h][k] - m);
      wgt[h][k] = e; ssum += e;
    }
    float inv = 1.f / ssum;
    for (int k = 0; k < NOFF; ++k) wgt[h][k] *= inv;

    const float* fqh = &fq[h * D_];
    const float* fsh = fksum + (size_t)(b * H_ + h) * D_;
    float zv = 0.f;
    for (int dd = 0; dd < D_; ++dd) zv += fqh[dd] * fsh[dd];
    zz[h] = zv;
  }
  __syncthreads();

  const int c0 = tid * 4;
  const int h  = c0 >> 7;
  const int e  = c0 & (D_ - 1);
  float out4[4] = {0.f, 0.f, 0.f, 0.f};

  // telescoping attention output
#pragma unroll 1
  for (int kk = 0; kk < NOFF; ++kk) {
    float w = wgt[h][kk];
    int off = (kk == 0) ? 0 : (1 << (kk - 1));
    int pos = s - off; if (pos < 0) pos = 0;   // weight is 0 when clipped
    const float* vr = qkv + (size_t)(b * S_ + pos) * N3_ + 2 * C_ + c0;
#pragma unroll
    for (int j = 0; j < 4; ++j) out4[j] += w * vr[j];
  }

  // causal depthwise conv, taps j=0..6 hit v[s-6+j]
#pragma unroll
  for (int j = 0; j < KC_; ++j) {
    int ps = s - (KC_ - 1) + j;
    if (ps >= 0) {
      const float* vr = qkv + (size_t)(b * S_ + ps) * N3_ + 2 * C_ + c0;
#pragma unroll
      for (int q = 0; q < 4; ++q) out4[q] += vr[q] * conv_w[(c0 + q) * KC_ + j];
    }
  }

  // linear attention: fq @ kv / (z + 1e-6)
  {
    const float* kvh = kvbuf + (size_t)(b * H_ + h) * D_ * D_ + e;
    const float* fqh = &fq[h * D_];
    float zinv = 1.f / (zz[h] + 1e-6f);
    float l4[4] = {0.f, 0.f, 0.f, 0.f};
#pragma unroll 4
    for (int dd = 0; dd < D_; ++dd) {
      float f = fqh[dd];
      const float* kr = kvh + (size_t)dd * D_;
#pragma unroll
      for (int j = 0; j < 4; ++j) l4[j] += f * kr[j];
    }
#pragma unroll
    for (int j = 0; j < 4; ++j) out4[j] += l4[j] * zinv;
  }

  bf16_t* op = hb + (size_t)(b * S_ + s) * C_ + c0;
#pragma unroll
  for (int j = 0; j < 4; ++j) op[j] = (bf16_t)out4[j];
}

// ---------------- launch ----------------
extern "C" void kernel_launch(void* const* d_in, const int* in_sizes, int n_in,
                              void* d_out, int out_size, void* d_ws, size_t ws_size,
                              hipStream_t stream) {
  (void)in_sizes; (void)n_in; (void)out_size; (void)ws_size;
  const float* x      = (const float*)d_in[0];
  const float* Wqkv   = (const float*)d_in[1];
  const float* Wo     = (const float*)d_in[2];
  const float* conv_w = (const float*)d_in[3];
  float* out = (float*)d_out;

  char* ws = (char*)d_ws;
  size_t off = 0;
  auto alloc = [&](size_t bytes) -> void* {
    off = (off + 255) & ~(size_t)255;
    void* p = ws + off;
    off += bytes;
    return p;
  };

  float*  qkv   = (float*) alloc((size_t)M_ * N3_ * 4);        // 96 MB
  bf16_t* xb    = (bf16_t*)alloc((size_t)M_ * C_ * 2);         // 16 MB
  bf16_t* WqkvT = (bf16_t*)alloc((size_t)N3_ * C_ * 2);        //  6 MB
  bf16_t* WoT   = (bf16_t*)alloc((size_t)C_ * C_ * 2);         //  2 MB
  bf16_t* hb    = (bf16_t*)alloc((size_t)M_ * C_ * 2);         // 16 MB
  float*  kvbuf = (float*) alloc((size_t)B_ * H_ * D_ * D_ * 4); // 1 MB
  float*  fksum = (float*) alloc((size_t)B_ * H_ * D_ * 4);

  (void)hipMemsetAsync(kvbuf, 0, (size_t)B_ * H_ * D_ * D_ * 4, stream);
  (void)hipMemsetAsync(fksum, 0, (size_t)B_ * H_ * D_ * 4, stream);

  k_f32_to_bf16   <<<1024, 256, 0, stream>>>(x, xb, M_ * C_);
  k_transpose_bf16<<<1024, 256, 0, stream>>>(Wqkv, WqkvT, C_, N3_);
  k_transpose_bf16<<<512,  256, 0, stream>>>(Wo,   WoT,   C_, C_);

  // qkv = x @ Wqkv (WMMA bf16, f32 accum)
  k_gemm_bf16_wmma<<<dim3(M_ / TM, N3_ / TN), 256, 0, stream>>>(
      xb, WqkvT, nullptr, qkv, M_, N3_, C_);

  // linear-attention state
  k_kvstats<<<dim3(B_ * H_, KV_NCH), 256, 0, stream>>>(qkv, kvbuf, fksum);

  // tele + conv + low  ->  bf16 h
  k_fuse<<<B_ * S_, 256, 0, stream>>>(qkv, kvbuf, fksum, conv_w, hb);

  // out = x + h @ Wo (WMMA bf16, f32 residual epilogue)
  k_gemm_bf16_wmma<<<dim3(M_ / TM, C_ / TN), 256, 0, stream>>>(
      hb, WoT, x, out, M_, C_, C_);
}
